// ScaledDotAttention_67826123538546
// MI455X (gfx1250) — matrix operand order repaired and checked by reading, not compiled, not run on hardware
//
#include <hip/hip_runtime.h>
#include <hip/hip_bf16.h>
#include <math.h>

typedef __bf16 bf16_t;
typedef __bf16 v16bf __attribute__((ext_vector_type(16)));
typedef __bf16 v8bf  __attribute__((ext_vector_type(8)));
typedef float  v8f   __attribute__((ext_vector_type(8)));
typedef float  v4f   __attribute__((ext_vector_type(4)));

#define DEVI __device__ __forceinline__

constexpr int    BB   = 8;
constexpr int    LL   = 2048;
constexpr int    DDIM = 256;
constexpr int    HHH  = 128;
constexpr size_t NROW = (size_t)BB * LL;   // 16384

// ---- workspace layout (byte offsets, base assumed 256B aligned) ----
constexpr size_t OFF_PQ  = 0;                                   // bf16 [NROW,H]
constexpr size_t OFF_PK  = OFF_PQ  + NROW * HHH * 2;            // bf16 [NROW,H]
constexpr size_t OFF_V1T = OFF_PK  + NROW * HHH * 2;            // bf16 [B][D,L]
constexpr size_t OFF_V2T = OFF_V1T + (size_t)BB * DDIM * LL * 2;
constexpr size_t OFF_WQB = OFF_V2T + (size_t)BB * DDIM * LL * 2;// bf16 [H,D]
constexpr size_t OFF_WKB = OFF_WQB + (size_t)HHH * DDIM * 2;
constexpr size_t OFF_M1  = OFF_WKB + (size_t)HHH * DDIM * 2;    // f32 [B*L]
constexpr size_t OFF_L1  = OFF_M1  + NROW * 4;
constexpr size_t OFF_M2  = OFF_L1  + NROW * 4;
constexpr size_t OFF_L2  = OFF_M2  + NROW * 4;
constexpr size_t OFF_S   = (OFF_L2 + NROW * 4 + 255) & ~(size_t)255; // f32 [B][L,L] 128MB

// ---- helpers ----
DEVI v8bf ld8bf(const bf16_t* p) { return *(const v8bf*)p; }
DEVI v4f  ld4f (const float*  p) { return *(const v4f*)p; }
DEVI v16bf cat16(v8bf lo, v8bf hi) {
  return __builtin_shufflevector(lo, hi, 0,1,2,3,4,5,6,7,8,9,10,11,12,13,14,15);
}
DEVI v8f wmma_bf16(v16bf a, v16bf b, v8f c) {
  return __builtin_amdgcn_wmma_f32_16x16x32_bf16(false, a, false, b, (short)0, c,
                                                 false, false);
}
// A-fragment K index for element j (0..15), given ks = 8*(lane>=16)
DEVI int kkidx(int j, int ks) { return (j < 8) ? (ks + j) : (16 + ks + (j - 8)); }

// async copy of 16B from global (saddr + 32-bit voffset, GVS mode) into LDS
// at workgroup LDS byte offset `ldsoff`; tracked by ASYNCcnt.
DEVI void async_g2l_b128(unsigned ldsoff, int voff, const void* sbase) {
  asm volatile("global_load_async_to_lds_b128 %0, %1, %2"
               :: "v"(ldsoff), "v"(voff), "s"(sbase) : "memory");
}
template <int N> DEVI void wait_asynccnt() {
  asm volatile("s_wait_asynccnt %0" :: "i"(N) : "memory");
}

// ---- kernel 1: W f32 -> bf16 ----
__global__ void cvt_w_kernel(const float* __restrict__ wq, const float* __restrict__ wk,
                             bf16_t* __restrict__ wqb, bf16_t* __restrict__ wkb) {
  int i = blockIdx.x * 256 + threadIdx.x;
  if (i < HHH * DDIM) { wqb[i] = (bf16_t)wq[i]; wkb[i] = (bf16_t)wk[i]; }
}

// ---- kernel 2: V [B][L,D] f32 -> VT [B][D,L] bf16 (LDS 32x32 tiles) ----
__global__ void vtrans_kernel(const float* __restrict__ v, bf16_t* __restrict__ vt) {
  __shared__ float tile[32][33];
  int b = blockIdx.z;
  int l0 = blockIdx.x * 32, d0 = blockIdx.y * 32;
  const float* vb = v + (size_t)b * LL * DDIM;
  bf16_t* vtb = vt + (size_t)b * DDIM * LL;
  #pragma unroll
  for (int i = 0; i < 4; ++i) {
    int l = threadIdx.y + i * 8;
    tile[l][threadIdx.x] = vb[(size_t)(l0 + l) * DDIM + d0 + threadIdx.x];
  }
  __syncthreads();
  #pragma unroll
  for (int i = 0; i < 4; ++i) {
    int dd = threadIdx.y + i * 8;
    vtb[(size_t)(d0 + dd) * LL + l0 + threadIdx.x] = (bf16_t)tile[threadIdx.x][dd];
  }
}

// ---- kernel 3: P = relu(X @ W^T) (* scale_h)  -> bf16 [NROW,H] ----
__global__ void proj_kernel(const float* __restrict__ X, const bf16_t* __restrict__ Wb,
                            const float* __restrict__ scale, bf16_t* __restrict__ P) {
  int wave = (blockIdx.x * blockDim.x + threadIdx.x) >> 5;
  int lane = threadIdx.x & 31, half = lane >> 4, lr = lane & 15, ks = half * 8;
  int m0 = wave * 16;
  if (m0 >= (int)NROW) return;
  const float* arow = X + (size_t)(m0 + lr) * DDIM;
  v8f acc[8] = {};
  for (int kc = 0; kc < DDIM / 32; ++kc) {
    const float* ap = arow + kc * 32;
    v4f x0 = ld4f(ap + ks),      x1 = ld4f(ap + ks + 4);
    v4f x2 = ld4f(ap + 16 + ks), x3 = ld4f(ap + 16 + ks + 4);
    v16bf af;
    #pragma unroll
    for (int j = 0; j < 4; ++j) {
      af[j]      = (bf16_t)x0[j];  af[4 + j]  = (bf16_t)x1[j];
      af[8 + j]  = (bf16_t)x2[j];  af[12 + j] = (bf16_t)x3[j];
    }
    #pragma unroll
    for (int t = 0; t < 8; ++t) {
      const bf16_t* bp = Wb + (size_t)(t * 16 + lr) * DDIM + kc * 32 + half * 8;
      acc[t] = wmma_bf16(af, cat16(ld8bf(bp), ld8bf(bp + 16)), acc[t]);
    }
  }
  #pragma unroll
  for (int t = 0; t < 8; ++t) {
    int n = t * 16 + lr;
    float sc = scale ? scale[n] : 1.0f;
    #pragma unroll
    for (int r = 0; r < 8; ++r) {
      float vv = acc[t][r];
      vv = (vv > 0.0f ? vv : 0.0f) * sc;
      P[(size_t)(m0 + r + half * 8) * HHH + n] = (bf16_t)vv;
    }
  }
}

// ---- kernel 4: S[b,q,k] = pq . pk (K = H = 128); wave tile 16q x 64k ----
__global__ void scores_kernel(const bf16_t* __restrict__ PQ, const bf16_t* __restrict__ PK,
                              float* __restrict__ S) {
  int wave = (blockIdx.x * blockDim.x + threadIdx.x) >> 5;
  int lane = threadIdx.x & 31, half = lane >> 4, lr = lane & 15, ks = half * 8;
  int kg = wave & 31, tmp = wave >> 5, qt = tmp & 127, b = tmp >> 7;
  int q0 = qt * 16, k0 = kg * 64;
  const bf16_t* pq = PQ + ((size_t)b * LL + q0 + lr) * HHH;
  v8f acc[4] = {};
  for (int kc = 0; kc < HHH / 32; ++kc) {
    const bf16_t* ap = pq + kc * 32;
    v16bf af = cat16(ld8bf(ap + ks), ld8bf(ap + 16 + ks));
    #pragma unroll
    for (int t = 0; t < 4; ++t) {
      const bf16_t* bp = PK + ((size_t)b * LL + k0 + t * 16 + lr) * HHH + kc * 32 + half * 8;
      acc[t] = wmma_bf16(af, cat16(ld8bf(bp), ld8bf(bp + 16)), acc[t]);
    }
  }
  float* Sb = S + (size_t)b * LL * LL;
  #pragma unroll
  for (int t = 0; t < 4; ++t)
    #pragma unroll
    for (int r = 0; r < 8; ++r)
      Sb[(size_t)(q0 + r + half * 8) * LL + k0 + t * 16 + lr] = acc[t][r];
}

// ---- kernel 5: per-row masked online max/sum (branch 1), wave per row ----
__global__ void row_stats_kernel(const float* __restrict__ S, const int* __restrict__ mask,
                                 float* __restrict__ m_out, float* __restrict__ l_out) {
  int wave = (blockIdx.x * blockDim.x + threadIdx.x) >> 5;
  int lane = threadIdx.x & 31;
  int b = wave >> 11, q = wave & 2047;
  const float* row = S + ((size_t)b * LL + q) * LL;
  const int* mk = mask + (size_t)b * LL;
  float m = -INFINITY, l = 0.0f;
  for (int k = lane; k < LL; k += 32) {
    if (!mk[k]) {
      float x = row[k];
      if (x > m) { l = l * __expf(m - x) + 1.0f; m = x; }
      else        l += __expf(x - m);
    }
  }
  #pragma unroll
  for (int off = 16; off > 0; off >>= 1) {
    float mo = __shfl_xor(m, off, 32);
    float lo = __shfl_xor(l, off, 32);
    float mn = fmaxf(m, mo);
    float la = (m == mn)  ? l  : l  * __expf(m  - mn);
    float lb = (mo == mn) ? lo : lo * __expf(mo - mn);
    l = la + lb; m = mn;
  }
  if (lane == 0) { m_out[wave] = m; l_out[wave] = l; }
}

// ---- kernel 6: per-column masked online max/sum (branch 2), thread per col ----
__global__ void col_stats_kernel(const float* __restrict__ S, const int* __restrict__ mask,
                                 float* __restrict__ m_out, float* __restrict__ l_out) {
  int g = blockIdx.x * blockDim.x + threadIdx.x;   // g = b*2048 + k
  int b = g >> 11, k = g & 2047;
  const float* Sb = S + (size_t)b * LL * LL;
  const int* mk = mask + (size_t)b * LL;
  float m = -INFINITY, l = 0.0f;
  for (int q = 0; q < LL; ++q) {
    if ((q & 15) == 0 && q + 16 < LL)
      __builtin_prefetch(&Sb[(size_t)(q + 16) * LL + k], 0, 0);  // global_prefetch
    if (!mk[q]) {
      float x = Sb[(size_t)q * LL + k];
      if (x > m) { l = l * __expf(m - x) + 1.0f; m = x; }
      else        l += __expf(x - m);
    }
  }
  m_out[g] = m; l_out[g] = l;
}

// ---- kernel 7: Out[m,:] = softmax-row(A) @ VT^T ; wave tile 16m x 128d ----
// B operand (VT 128-row x 32-col bf16 chunk, 8KB) is double-buffered in LDS via
// GLOBAL_LOAD_ASYNC_TO_LDS_B128 (ASYNCcnt), consumed with ds_load_b128.
// TRANS=0: A row m = S row (branch 1).  TRANS=1: A row m = S column (branch 2).
template <int TRANS>
__global__ void attn_out_kernel(const float* __restrict__ S,
                                const float* __restrict__ m_arr,
                                const float* __restrict__ l_arr,
                                const int* __restrict__ cmask,
                                const bf16_t* __restrict__ VT,
                                float* __restrict__ Out) {
  extern __shared__ char smem_raw[];     // 8 waves * 2 bufs * 8KB = 128KB
  int wave = (blockIdx.x * blockDim.x + threadIdx.x) >> 5;
  int wib  = threadIdx.x >> 5;
  int lane = threadIdx.x & 31, half = lane >> 4, lr = lane & 15, ks = half * 8;
  int dg = wave & 1, tmp = wave >> 1, mt = tmp & 127, b = tmp >> 7;
  int m0 = mt * 16, d0 = dg * 128;
  const float* Sb = S + (size_t)b * LL * LL;
  const float* mb = m_arr + (size_t)b * LL;
  const float* lb = l_arr + (size_t)b * LL;
  const int*   cm = cmask + (size_t)b * LL;
  const bf16_t* VTb = VT + (size_t)b * DDIM * LL;
  float mr = mb[m0 + lr];            // max of this lane's A row

  unsigned lbase = (unsigned)(wib * 16384);          // this wave's LDS region
  // per-lane async-copy addressing: 16 issues x 32 lanes x 16B cover 128x32 bf16
  int rbase = d0 + (lane >> 2);                      // VT row for issue 0
  int cg16  = (lane & 3) * 16;                       // byte offset within 64B row

  auto issue_chunk = [&](int kc, int buf) {
    unsigned lo = lbase + (unsigned)buf * 8192u + (unsigned)(lane * 16);
    int voff = rbase * (LL * 2) + cg16 + kc * 64;
    #pragma unroll
    for (int i = 0; i < 16; ++i) {
      async_g2l_b128(lo, voff, VTb);
      lo += 512u;                 // 32 lanes * 16B
      voff += 8 * (LL * 2);       // 8 VT rows per issue
    }
  };

  v8f acc[8] = {};
  issue_chunk(0, 0);
  for (int kc = 0; kc < LL / 32; ++kc) {
    int cur = kc & 1;
    if (kc + 1 < LL / 32) {
      issue_chunk(kc + 1, cur ^ 1);
      wait_asynccnt<16>();        // chunk kc complete (in-order), kc+1 in flight
    } else {
      wait_asynccnt<0>();
    }
    float a[16];
    if (TRANS == 0) {
      const float* ap = Sb + (size_t)(m0 + lr) * LL + kc * 32;
      v4f x0 = ld4f(ap + ks),      x1 = ld4f(ap + ks + 4);
      v4f x2 = ld4f(ap + 16 + ks), x3 = ld4f(ap + 16 + ks + 4);
      #pragma unroll
      for (int j = 0; j < 4; ++j) {
        a[j] = x0[j]; a[4 + j] = x1[j]; a[8 + j] = x2[j]; a[12 + j] = x3[j];
      }
    } else {
      #pragma unroll
      for (int j = 0; j < 16; ++j)
        a[j] = Sb[(size_t)(kc * 32 + kkidx(j, ks)) * LL + (m0 + lr)];
    }
    v16bf af;
    #pragma unroll
    for (int j = 0; j < 16; ++j) {
      int col = kc * 32 + kkidx(j, ks);
      float e = cm[col] ? 0.0f : __expf(a[j] - mr);
      af[j] = (bf16_t)e;
    }
    const bf16_t* bbuf = (const bf16_t*)(smem_raw + wib * 16384 + cur * 8192);
    #pragma unroll
    for (int t = 0; t < 8; ++t) {
      const bf16_t* bp = bbuf + (t * 16 + lr) * 32 + half * 8;
      acc[t] = wmma_bf16(af, cat16(ld8bf(bp), ld8bf(bp + 16)), acc[t]);
    }
  }
  #pragma unroll
  for (int r = 0; r < 8; ++r) {
    int row = m0 + r + half * 8;
    float linv = 1.0f / lb[row];
    #pragma unroll
    for (int t = 0; t < 8; ++t)
      Out[((size_t)b * LL + row) * DDIM + d0 + t * 16 + lr] = acc[t][r] * linv;
  }
}

extern "C" void kernel_launch(void* const* d_in, const int* in_sizes, int n_in,
                              void* d_out, int out_size, void* d_ws, size_t ws_size,
                              hipStream_t stream) {
  (void)in_sizes; (void)n_in; (void)out_size; (void)ws_size;
  const float* Q   = (const float*)d_in[0];
  const float* K   = (const float*)d_in[1];
  const float* V1  = (const float*)d_in[2];
  const int*   Mk1 = (const int*)  d_in[3];   // values_1_mask [B,Lk]
  const float* V2  = (const float*)d_in[4];
  const int*   Mk2 = (const int*)  d_in[5];   // values_2_mask [B,Lq]
  const float* Wq  = (const float*)d_in[6];
  const float* Wk  = (const float*)d_in[7];
  const float* Sc  = (const float*)d_in[8];   // scaling [1,H]

  char* ws = (char*)d_ws;
  bf16_t* pq  = (bf16_t*)(ws + OFF_PQ);
  bf16_t* pk  = (bf16_t*)(ws + OFF_PK);
  bf16_t* v1t = (bf16_t*)(ws + OFF_V1T);
  bf16_t* v2t = (bf16_t*)(ws + OFF_V2T);
  bf16_t* wqb = (bf16_t*)(ws + OFF_WQB);
  bf16_t* wkb = (bf16_t*)(ws + OFF_WKB);
  float*  m1  = (float*) (ws + OFF_M1);
  float*  l1  = (float*) (ws + OFF_L1);
  float*  m2  = (float*) (ws + OFF_M2);
  float*  l2  = (float*) (ws + OFF_L2);
  float*  S   = (float*) (ws + OFF_S);

  float* out1 = (float*)d_out;                         // [B,Lq,D]
  float* out2 = out1 + (size_t)BB * LL * DDIM;         // [B,Lk,D]

  cvt_w_kernel<<<(HHH * DDIM + 255) / 256, 256, 0, stream>>>(Wq, Wk, wqb, wkb);

  dim3 tg(LL / 32, DDIM / 32, BB), tb(32, 8);
  vtrans_kernel<<<tg, tb, 0, stream>>>(V1, v1t);
  vtrans_kernel<<<tg, tb, 0, stream>>>(V2, v2t);

  // 1024 waves each (16 rows/wave over 16384 rows), 8 waves/block
  proj_kernel<<<128, 256, 0, stream>>>(Q, wqb, nullptr, pq);
  proj_kernel<<<128, 256, 0, stream>>>(K, wkb, Sc, pk);

  // 8 * 128 * 32 = 32768 waves
  scores_kernel<<<4096, 256, 0, stream>>>(pq, pk, S);

  row_stats_kernel<<<2048, 256, 0, stream>>>(S, Mk1, m1, l1);   // 16384 waves
  col_stats_kernel<<<64,   256, 0, stream>>>(S, Mk2, m2, l2);   // 16384 threads

  // 8 * 128 * 2 = 2048 waves each; 128KB dynamic LDS (8 waves * 2 * 8KB)
  attn_out_kernel<0><<<256, 256, 131072, stream>>>(S, m1, l1, Mk1, v1t, out1);
  attn_out_kernel<1><<<256, 256, 131072, stream>>>(S, m2, l2, Mk2, v2t, out2);
}
